// MLoss_72773925864255
// MI455X (gfx1250) — compile-verified, matrix operand order checked by reading, not run on hardware
//
#include <hip/hip_runtime.h>

typedef float  v4f  __attribute__((ext_vector_type(4)));
typedef float  v8f  __attribute__((ext_vector_type(8)));
typedef _Float16 v16h __attribute__((ext_vector_type(16)));

#define IMG_INV  (1.0f / 416.0f)
#define THRESH   0.5f
#define ALPHA    0.1f

__device__ __forceinline__ float fast_sigmoid(float v) {
    // 1 / (1 + e^-v): v_exp_f32 + rcp
    return __frcp_rn(1.0f + __expf(-v));
}

// Kernel 1: streaming masked-squared-error partial sums.
// Each thread: 4 cells = 20 floats per array = 5 x b128 NT loads per array.
// Wave reduction via v_wmma_f32_16x16x32_f16 (exact: 3-term f16 expansion of
// each lane's f32 partial, B = ones, f32 accumulate), finished with 7 adds
// and one xor-16 shuffle. Block partial -> d_ws (deterministic).
__global__ void __launch_bounds__(256)
mloss_partial_kernel(const float* __restrict__ x, const float* __restrict__ y,
                     float* __restrict__ partial, int totalCells) {
    const int tid = blockIdx.x * 256 + threadIdx.x;
    const int cellBase = tid * 4;

    float acc = 0.0f;
    if (cellBase < totalCells) {
        const v4f* px = (const v4f*)(x + (size_t)cellBase * 5);
        const v4f* py = (const v4f*)(y + (size_t)cellBase * 5);
        float xa[20], ya[20];
#pragma unroll
        for (int i = 0; i < 5; ++i) {
            v4f vx = __builtin_nontemporal_load(px + i);   // global_load_b128, TH=NT
            v4f vy = __builtin_nontemporal_load(py + i);
#pragma unroll
            for (int j = 0; j < 4; ++j) { xa[4 * i + j] = vx[j]; ya[4 * i + j] = vy[j]; }
        }
#pragma unroll
        for (int j = 0; j < 4; ++j) {
            const float* cx = &xa[5 * j];
            const float* cy = &ya[5 * j];
            float o0 = fast_sigmoid(cx[0]);
            float o1 = fast_sigmoid(cx[1]);
            float o2 = __expf(cx[2]) * IMG_INV;
            float o3 = __expf(cx[3]) * IMG_INV;
            float o4 = fast_sigmoid(cx[4]);
            float d0 = cy[0] - o0, d1 = cy[1] - o1, d2 = cy[2] - o2;
            float d3 = cy[3] - o3, d4 = cy[4] - o4;
            float sq4 = d4 * d4;
            float s5  = fmaf(d0, d0, fmaf(d1, d1, fmaf(d2, d2, fmaf(d3, d3, sq4))));
            acc += (cy[4] > THRESH) ? s5 : ALPHA * sq4;
        }
    }

    // ---- exact 3-term f16 expansion of the f32 lane partial ----
    _Float16 h0 = (_Float16)acc;
    float r = acc - (float)h0;
    _Float16 h1 = (_Float16)r;
    r = r - (float)h1;
    _Float16 h2 = (_Float16)r;

    v16h a = {};
    a[0] = h0; a[1] = h1; a[2] = h2;
    v16h bones;
#pragma unroll
    for (int i = 0; i < 16; ++i) bones[i] = (_Float16)1.0f;

    // D[m][n] = sum_k A[m][k] * 1  (+0) => t[m] = s[m] + s[m+16] in f32
    v8f c = {};
    c = __builtin_amdgcn_wmma_f32_16x16x32_f16(
        /*neg_a=*/false, a, /*neg_b=*/false, bones,
        /*c_mod=*/(short)0, c, /*reuse_a=*/false, /*reuse_b=*/false);

    // lanes 0-15 hold t[0..7] in c[0..7]; lanes 16-31 hold t[8..15]
    float u = ((c[0] + c[1]) + (c[2] + c[3])) + ((c[4] + c[5]) + (c[6] + c[7]));
    u += __shfl_xor(u, 16, 32);   // wave total on every lane

    __shared__ float wsum[8];
    const int lane = threadIdx.x & 31;
    const int wid  = threadIdx.x >> 5;
    if (lane == 0) wsum[wid] = u;
    __syncthreads();
    if (threadIdx.x == 0) {
        float b = 0.0f;
#pragma unroll
        for (int i = 0; i < 8; ++i) b += wsum[i];   // fixed order -> deterministic
        partial[blockIdx.x] = b;
    }
}

// Kernel 2: deterministic fixed-order reduction of block partials + mean.
__global__ void __launch_bounds__(256)
mloss_final_kernel(const float* __restrict__ partial, int n,
                   float invDivisor, float* __restrict__ out) {
    __shared__ float sm[256];
    float s = 0.0f;
    for (int i = threadIdx.x; i < n; i += 256) s += partial[i];
    sm[threadIdx.x] = s;
    __syncthreads();
#pragma unroll
    for (int ofs = 128; ofs > 0; ofs >>= 1) {
        if (threadIdx.x < ofs) sm[threadIdx.x] += sm[threadIdx.x + ofs];
        __syncthreads();
    }
    if (threadIdx.x == 0) out[0] = sm[0] * invDivisor;
}

extern "C" void kernel_launch(void* const* d_in, const int* in_sizes, int n_in,
                              void* d_out, int out_size, void* d_ws, size_t ws_size,
                              hipStream_t stream) {
    const float* x = (const float*)d_in[0];   // [256, 10647, 5] f32
    const float* y = (const float*)d_in[1];   // [256, 10647, 5] f32
    float* ws = (float*)d_ws;

    const int totalCells = in_sizes[0] / 5;             // 2,725,632
    const int threads    = (totalCells + 3) / 4;        // 4 cells / thread
    const int blocks     = (threads + 255) / 256;       // 2662

    // mean over [B, 5]: divisor = B*5 = total_elems / N(=10647)
    const float invDivisor = 10647.0f / (float)in_sizes[0];   // = 1/1280

    mloss_partial_kernel<<<blocks, 256, 0, stream>>>(x, y, ws, totalCells);
    mloss_final_kernel<<<1, 256, 0, stream>>>(ws, blocks, invDivisor, (float*)d_out);
}
